// GCNClusterGAT_16303695856290
// MI455X (gfx1250) — matrix operand order repaired and checked by reading, not compiled, not run on hardware
//
#include <hip/hip_runtime.h>
#include <hip/hip_bf16.h>
#include <math.h>

#define NN     20000
#define EE     640000
#define EPRIME 660000          // EE + NN self loops
#define NFEAT  512
#define NHID   128
#define HEADS  2
#define C1     (HEADS*NHID)    // 256
#define NOUT   64
#define KCLUS  100
#define SLOPE  0.2f
#define CTEMP  5.0f

typedef __attribute__((ext_vector_type(16))) __bf16 v16bf;
typedef __attribute__((ext_vector_type(8)))  float  v8f;

// ---------------------------------------------------------------- helpers
__device__ __forceinline__ unsigned short f32_to_bf16_rne(float f) {
  unsigned u = __float_as_uint(f);
  unsigned r = 0x7FFFu + ((u >> 16) & 1u);
  return (unsigned short)((u + r) >> 16);
}

__device__ __forceinline__ void atomic_max_f32(float* addr, float val) {
  // monotone-encoding trick; storage initialized to -inf
  if (val >= 0.f) atomicMax((int*)addr, __float_as_int(val));
  else            atomicMin((unsigned int*)addr, __float_as_uint(val));
}

__device__ __forceinline__ void edge_sd(const int* __restrict__ ei, int e,
                                        int& s, int& d) {
  if (e < EE) { s = ei[e]; d = ei[EE + e]; }
  else        { s = e - EE; d = e - EE; }
}

// ---------------------------------------------------------------- utility
__global__ void cvt_bf16(const float* __restrict__ in,
                         unsigned short* __restrict__ out, int n) {
  int i = blockIdx.x * blockDim.x + threadIdx.x;
  if (i < n) out[i] = f32_to_bf16_rne(in[i]);
}

__global__ void fill_f32(float* __restrict__ p, float v, int n) {
  int i = blockIdx.x * blockDim.x + threadIdx.x;
  if (i < n) p[i] = v;
}

// ---------------------------------------------------------------- WMMA GEMM
// C[M,N](f32) = A[M,K](bf16, row major) * W[N,K](bf16, row major)^T
// block = 128 threads (4 waves). blockIdx.x -> M/16 tile, blockIdx.y -> N/64.
// The whole 16xK A strip (<=16KB bf16) is staged to LDS ONCE via the CDNA5
// async memory->LDS path (ASYNCcnt), then the hot loop is pure
// ds_load_b128 + global_load_b128 + v_wmma with no barriers.
__global__ __launch_bounds__(128) void gemm_wmma_bf16(
    const unsigned short* __restrict__ A,
    const unsigned short* __restrict__ W,
    float* __restrict__ C, int M, int N, int K) {
  __shared__ unsigned As[16 * 256];             // up to 16 rows x 512 bf16
  const unsigned* A32 = (const unsigned*)A;
  const unsigned* W32 = (const unsigned*)W;
  int wave = threadIdx.x >> 5;
  int lane = threadIdx.x & 31;
  int m0 = blockIdx.x * 16;
  int n0 = blockIdx.y * 64 + wave * 16;
  int mloc = lane & 15;
  int hi   = lane >> 4;
  int Kw = K >> 1;                              // u32 per row

  // ---- one-shot async stage of the 16xK A strip (uint2 granules) ----
  int total = 16 * Kw;                          // u32 elements
  for (int i = threadIdx.x * 2; i < total; i += 256) {
    int row = i / Kw;
    int col = i - row * Kw;
    const unsigned* gp = A32 + (size_t)(m0 + row) * Kw + col;
    unsigned ldsoff = (unsigned)(uintptr_t)(&As[i]);   // flat addr[31:0] == LDS offset
    asm volatile("global_load_async_to_lds_b64 %0, %1, off"
                 :: "v"(ldsoff), "v"(gp) : "memory");
  }
  asm volatile("s_wait_asynccnt 0" ::: "memory");
  __syncthreads();

  // ---- hot loop: 1 WMMA per 32-wide K chunk, no barriers ----
  v8f acc = {0.f, 0.f, 0.f, 0.f, 0.f, 0.f, 0.f, 0.f};
  int n = n0 + mloc;
  for (int kk = 0; kk < K; kk += 32) {
    union { v16bf v; unsigned u[8]; } af, bfr;
    #pragma unroll
    for (int i = 0; i < 8; i++) {               // A frag per ISA table
      int k0 = (i < 4 ? 2 * i : 16 + 2 * (i - 4)) + hi * 8;
      af.u[i] = As[mloc * Kw + (kk >> 1) + (k0 >> 1)];
    }
    #pragma unroll
    for (int i = 0; i < 8; i++) {               // B frag: B[k][n] = W[n][k]
      int k = kk + (i < 4 ? 2 * i : 16 + 2 * (i - 4)) + hi * 8;
      bfr.u[i] = W32[(size_t)n * Kw + (k >> 1)];
    }
    acc = __builtin_amdgcn_wmma_f32_16x16x32_bf16(
        false, af.v, false, bfr.v, (short)0, acc, false, false);
  }
  #pragma unroll
  for (int i = 0; i < 8; i++) {                 // D layout: m = i + 8*hi
    int m = m0 + i + 8 * hi;
    C[(size_t)m * N + n] = acc[i];
  }
}

// ---------------------------------------------------------------- GAT alphas
__global__ __launch_bounds__(256) void alpha1_kernel(
    const float* __restrict__ h, const float* __restrict__ a_src,
    const float* __restrict__ a_dst, float* __restrict__ as,
    float* __restrict__ ad) {
  int wid  = (blockIdx.x * blockDim.x + threadIdx.x) >> 5;
  int lane = threadIdx.x & 31;
  if (wid >= NN) return;
  const float* row = h + (size_t)wid * C1;
  int base = lane * 8;                          // lanes 0-15 head0, 16-31 head1
  float ps = 0.f, pd = 0.f;
  #pragma unroll
  for (int j = 0; j < 8; j++) {
    float v = row[base + j];
    ps += v * a_src[base + j];
    pd += v * a_dst[base + j];
  }
  #pragma unroll
  for (int o = 1; o < 16; o <<= 1) {
    ps += __shfl_xor(ps, o);
    pd += __shfl_xor(pd, o);
  }
  if ((lane & 15) == 0) {
    int hh = lane >> 4;
    as[wid * 2 + hh] = ps;
    ad[wid * 2 + hh] = pd;
  }
}

__global__ __launch_bounds__(256) void alpha2_kernel(
    const float* __restrict__ h, const float* __restrict__ a_src,
    const float* __restrict__ a_dst, float* __restrict__ as,
    float* __restrict__ ad) {
  int wid  = (blockIdx.x * blockDim.x + threadIdx.x) >> 5;
  int lane = threadIdx.x & 31;
  if (wid >= NN) return;
  const float* row = h + (size_t)wid * NOUT;
  float v0 = row[lane * 2], v1 = row[lane * 2 + 1];
  float ps = v0 * a_src[lane * 2] + v1 * a_src[lane * 2 + 1];
  float pd = v0 * a_dst[lane * 2] + v1 * a_dst[lane * 2 + 1];
  #pragma unroll
  for (int o = 1; o < 32; o <<= 1) {
    ps += __shfl_xor(ps, o);
    pd += __shfl_xor(pd, o);
  }
  if (lane == 0) { as[wid] = ps; ad[wid] = pd; }
}

// ---------------------------------------------------------------- edge passes
__global__ void edge_logits(const int* __restrict__ ei,
                            const float* __restrict__ as,
                            const float* __restrict__ ad,
                            float* __restrict__ eb,
                            float* __restrict__ mmax, int H) {
  int idx = blockIdx.x * blockDim.x + threadIdx.x;
  if (idx >= EPRIME * H) return;
  int e = idx / H, hh = idx % H;
  int s, d; edge_sd(ei, e, s, d);
  float v = as[s * H + hh] + ad[d * H + hh];
  v = v > 0.f ? v : SLOPE * v;
  eb[idx] = v;
  atomic_max_f32(&mmax[d * H + hh], v);
}

__global__ void edge_exp(const int* __restrict__ ei, float* __restrict__ eb,
                         const float* __restrict__ mmax,
                         float* __restrict__ ssum, int H) {
  int idx = blockIdx.x * blockDim.x + threadIdx.x;
  if (idx >= EPRIME * H) return;
  int e = idx / H, hh = idx % H;
  int s, d; edge_sd(ei, e, s, d);
  (void)s;
  float ex = __expf(eb[idx] - mmax[d * H + hh]);
  eb[idx] = ex;
  atomicAdd(&ssum[d * H + hh], ex);
}

// layer 1: wave per (edge, head), 128 floats each
__global__ __launch_bounds__(256) void edge_agg1(
    const int* __restrict__ ei, const float* __restrict__ eb,
    const float* __restrict__ ssum, const float* __restrict__ h,
    float* __restrict__ agg) {
  int wid  = (blockIdx.x * blockDim.x + threadIdx.x) >> 5;
  int lane = threadIdx.x & 31;
  if (wid >= EPRIME * 2) return;
  int e = wid >> 1, hh = wid & 1;
  int s, d; edge_sd(ei, e, s, d);
  float coef = eb[wid] / (ssum[d * 2 + hh] + 1e-16f);
  const float4* src4 = (const float4*)(h + (size_t)s * C1 + hh * NHID);
  __builtin_prefetch(src4, 0, 3);
  float4 v = src4[lane];
  float* dst = agg + (size_t)d * C1 + hh * NHID + lane * 4;
  atomicAdd(dst + 0, v.x * coef);
  atomicAdd(dst + 1, v.y * coef);
  atomicAdd(dst + 2, v.z * coef);
  atomicAdd(dst + 3, v.w * coef);
}

// layer 2: wave per edge, 64 floats
__global__ __launch_bounds__(256) void edge_agg2(
    const int* __restrict__ ei, const float* __restrict__ eb,
    const float* __restrict__ ssum, const float* __restrict__ h,
    float* __restrict__ agg) {
  int wid  = (blockIdx.x * blockDim.x + threadIdx.x) >> 5;
  int lane = threadIdx.x & 31;
  if (wid >= EPRIME) return;
  int s, d; edge_sd(ei, wid, s, d);
  float coef = eb[wid] / (ssum[d] + 1e-16f);
  const float2* src2 = (const float2*)(h + (size_t)s * NOUT);
  float2 v = src2[lane];
  float* dst = agg + (size_t)d * NOUT + lane * 2;
  atomicAdd(dst + 0, v.x * coef);
  atomicAdd(dst + 1, v.y * coef);
}

// ---------------------------------------------------------------- finalize
__global__ void finalize1(const float* __restrict__ agg,
                          const float* __restrict__ b,
                          unsigned short* __restrict__ h1b) {
  int i = blockIdx.x * blockDim.x + threadIdx.x;
  if (i >= NN * C1) return;
  float v = agg[i] + b[i % C1];
  v = v > 0.f ? v : (__expf(v) - 1.f);          // ELU
  h1b[i] = f32_to_bf16_rne(v);
}

__global__ __launch_bounds__(256) void finalize2(
    const float* __restrict__ agg, const float* __restrict__ b,
    float* __restrict__ emb_out, float* __restrict__ data_n) {
  int wid  = (blockIdx.x * blockDim.x + threadIdx.x) >> 5;
  int lane = threadIdx.x & 31;
  if (wid >= NN) return;
  int c = lane * 2;
  size_t o = (size_t)wid * NOUT + c;
  float v0 = agg[o]     + b[c]     + 1e-6f;
  float v1 = agg[o + 1] + b[c + 1] + 1e-6f;
  emb_out[o] = v0; emb_out[o + 1] = v1;
  float ss = v0 * v0 + v1 * v1;
  #pragma unroll
  for (int off = 16; off > 0; off >>= 1) ss += __shfl_xor(ss, off);
  float rn = 1.0f / sqrtf(ss);
  data_n[o] = v0 * rn; data_n[o + 1] = v1 * rn;
}

// ---------------------------------------------------------------- clustering
// dist[N,K] = data_n @ mu^T ; block handles 32 nodes, thread t<100 owns mu row t
__global__ __launch_bounds__(128) void dist_kernel(
    const float* __restrict__ data_n, const float* __restrict__ mu,
    float* __restrict__ dist) {
  __shared__ float ds[32 * 64];
  int n0 = blockIdx.x * 32;
  const float4* g = (const float4*)(data_n + (size_t)n0 * NOUT);
  float4* s4 = (float4*)ds;
  for (int i = threadIdx.x; i < 32 * 16; i += 128) s4[i] = g[i];
  __syncthreads();
  if (threadIdx.x < KCLUS) {
    float murow[NOUT];
    #pragma unroll
    for (int d = 0; d < NOUT; d++) murow[d] = mu[threadIdx.x * NOUT + d];
    for (int nl = 0; nl < 32; nl++) {
      float acc = 0.f;
      #pragma unroll
      for (int d = 0; d < NOUT; d++) acc += ds[nl * 64 + d] * murow[d];
      dist[(size_t)(n0 + nl) * KCLUS + threadIdx.x] = acc;
    }
  }
}

// wave per node: softmax(CTEMP*dist) row -> r; optional atomic cluster_r accum
__global__ __launch_bounds__(256) void softmax_rows(
    const float* __restrict__ dist, float* __restrict__ r,
    float* __restrict__ cluster_r) {
  int wid  = (blockIdx.x * blockDim.x + threadIdx.x) >> 5;
  int lane = threadIdx.x & 31;
  if (wid >= NN) return;
  const float* dr = dist + (size_t)wid * KCLUS;
  float v0 = dr[lane], v1 = dr[32 + lane], v2 = dr[64 + lane];
  float v3 = (lane < 4) ? dr[96 + lane] : -__builtin_inff();
  float m = fmaxf(fmaxf(v0, v1), fmaxf(v2, v3));
  #pragma unroll
  for (int o = 16; o > 0; o >>= 1) m = fmaxf(m, __shfl_xor(m, o));
  float e0 = __expf(CTEMP * (v0 - m));
  float e1 = __expf(CTEMP * (v1 - m));
  float e2 = __expf(CTEMP * (v2 - m));
  float e3 = (lane < 4) ? __expf(CTEMP * (v3 - m)) : 0.f;
  float s = e0 + e1 + e2 + e3;
  #pragma unroll
  for (int o = 16; o > 0; o >>= 1) s += __shfl_xor(s, o);
  float inv = 1.0f / s;
  float r0 = e0 * inv, r1 = e1 * inv, r2 = e2 * inv, r3 = e3 * inv;
  float* rr = r + (size_t)wid * KCLUS;
  rr[lane] = r0; rr[32 + lane] = r1; rr[64 + lane] = r2;
  if (lane < 4) rr[96 + lane] = r3;
  if (cluster_r) {
    atomicAdd(&cluster_r[lane], r0);
    atomicAdd(&cluster_r[32 + lane], r1);
    atomicAdd(&cluster_r[64 + lane], r2);
    if (lane < 4) atomicAdd(&cluster_r[96 + lane], r3);
  }
}

// cluster_sum[K,NOUT] += r^T @ data_n  (split over 160 node chunks)
__global__ __launch_bounds__(256) void rtd_kernel(
    const float* __restrict__ r, const float* __restrict__ data_n,
    float* __restrict__ cluster_sum) {
  __shared__ float rs[KCLUS];
  __shared__ float dsrow[NOUT];
  float acc[25];
  #pragma unroll
  for (int j = 0; j < 25; j++) acc[j] = 0.f;
  const int chunk = NN / 160;                   // 125
  int nst = blockIdx.x * chunk, nen = nst + chunk;
  for (int n = nst; n < nen; n++) {
    __syncthreads();
    if (threadIdx.x < KCLUS)
      rs[threadIdx.x] = r[(size_t)n * KCLUS + threadIdx.x];
    else if (threadIdx.x < KCLUS + NOUT)
      dsrow[threadIdx.x - KCLUS] = data_n[(size_t)n * NOUT + threadIdx.x - KCLUS];
    __syncthreads();
    #pragma unroll
    for (int j = 0; j < 25; j++) {
      int c = threadIdx.x + 256 * j;            // c < 6400
      acc[j] += rs[c >> 6] * dsrow[c & 63];
    }
  }
  #pragma unroll
  for (int j = 0; j < 25; j++)
    atomicAdd(&cluster_sum[threadIdx.x + 256 * j], acc[j]);
}

__global__ void mu_update(const float* __restrict__ cluster_sum,
                          const float* __restrict__ cluster_r,
                          float* __restrict__ mu) {
  int i = blockIdx.x * blockDim.x + threadIdx.x;
  if (i < KCLUS * NOUT) mu[i] = cluster_sum[i] / cluster_r[i >> 6];
}

// ---------------------------------------------------------------- launch
extern "C" void kernel_launch(void* const* d_in, const int* in_sizes, int n_in,
                              void* d_out, int out_size, void* d_ws,
                              size_t ws_size, hipStream_t stream) {
  const float* x      = (const float*)d_in[0];
  const float* W1     = (const float*)d_in[1];
  const float* a_src1 = (const float*)d_in[2];
  const float* a_dst1 = (const float*)d_in[3];
  const float* b1     = (const float*)d_in[4];
  const float* W2     = (const float*)d_in[5];
  const float* a_src2 = (const float*)d_in[6];
  const float* a_dst2 = (const float*)d_in[7];
  const float* b2     = (const float*)d_in[8];
  const float* mu0    = (const float*)d_in[9];
  const int*   ei     = (const int*)d_in[10];
  (void)in_sizes; (void)n_in; (void)out_size; (void)ws_size;

  // ---- workspace carve-out (256B aligned) ----
  char* Wb = (char*)d_ws;
  size_t off = 0;
  auto nxt = [&](size_t bytes) -> void* {
    void* p = Wb + off;
    off += (bytes + 255) & ~(size_t)255;
    return p;
  };
  unsigned short* xb   = (unsigned short*)nxt((size_t)NN * NFEAT * 2);
  unsigned short* w1b  = (unsigned short*)nxt((size_t)C1 * NFEAT * 2);
  unsigned short* w2b  = (unsigned short*)nxt((size_t)NOUT * C1 * 2);
  float* h1pre = (float*)nxt((size_t)NN * C1 * 4);
  float* as1   = (float*)nxt((size_t)NN * 2 * 4);
  float* ad1   = (float*)nxt((size_t)NN * 2 * 4);
  float* m1    = (float*)nxt((size_t)NN * 2 * 4);
  float* s1    = (float*)nxt((size_t)NN * 2 * 4);
  float* e1b   = (float*)nxt((size_t)EPRIME * 2 * 4);
  float* agg1  = (float*)nxt((size_t)NN * C1 * 4);
  unsigned short* h1b = (unsigned short*)nxt((size_t)NN * C1 * 2);
  float* h2pre = (float*)nxt((size_t)NN * NOUT * 4);
  float* as2   = (float*)nxt((size_t)NN * 4);
  float* ad2   = (float*)nxt((size_t)NN * 4);
  float* m2    = (float*)nxt((size_t)NN * 4);
  float* s2    = (float*)nxt((size_t)NN * 4);
  float* e2b   = (float*)nxt((size_t)EPRIME * 4);
  float* agg2  = (float*)nxt((size_t)NN * NOUT * 4);
  float* datan = (float*)nxt((size_t)NN * NOUT * 4);
  float* mu    = (float*)nxt((size_t)KCLUS * NOUT * 4);
  float* crw   = (float*)nxt(512);
  float* csum  = (float*)nxt((size_t)KCLUS * NOUT * 4);
  float* distw = (float*)nxt((size_t)NN * KCLUS * 4);
  float* rw    = (float*)nxt((size_t)NN * KCLUS * 4);

  float* out      = (float*)d_out;
  float* out_mu   = out;                        // [100,64]
  float* out_r    = out + 6400;                 // [20000,100]
  float* out_emb  = out + 6400 + 2000000;       // [20000,64]
  float* out_dist = out + 6400 + 2000000 + 1280000;  // [20000,100]

  const float NEGINF = -__builtin_inff();

  // ---- bf16 conversions ----
  cvt_bf16<<<(NN * NFEAT + 255) / 256, 256, 0, stream>>>(x, xb, NN * NFEAT);
  cvt_bf16<<<(C1 * NFEAT + 255) / 256, 256, 0, stream>>>(W1, w1b, C1 * NFEAT);
  cvt_bf16<<<(NOUT * C1 + 255) / 256, 256, 0, stream>>>(W2, w2b, NOUT * C1);

  // ---- GAT layer 1 ----
  gemm_wmma_bf16<<<dim3(NN / 16, C1 / 64), 128, 0, stream>>>(
      xb, w1b, h1pre, NN, C1, NFEAT);
  alpha1_kernel<<<(NN * 32 + 255) / 256, 256, 0, stream>>>(
      h1pre, a_src1, a_dst1, as1, ad1);
  fill_f32<<<(NN * 2 + 255) / 256, 256, 0, stream>>>(m1, NEGINF, NN * 2);
  hipMemsetAsync(s1, 0, (size_t)NN * 2 * 4, stream);
  hipMemsetAsync(agg1, 0, (size_t)NN * C1 * 4, stream);
  edge_logits<<<(EPRIME * 2 + 255) / 256, 256, 0, stream>>>(ei, as1, ad1, e1b, m1, 2);
  edge_exp<<<(EPRIME * 2 + 255) / 256, 256, 0, stream>>>(ei, e1b, m1, s1, 2);
  edge_agg1<<<((size_t)EPRIME * 2 + 7) / 8, 256, 0, stream>>>(ei, e1b, s1, h1pre, agg1);
  finalize1<<<(NN * C1 + 255) / 256, 256, 0, stream>>>(agg1, b1, h1b);

  // ---- GAT layer 2 ----
  gemm_wmma_bf16<<<dim3(NN / 16, NOUT / 64), 128, 0, stream>>>(
      h1b, w2b, h2pre, NN, NOUT, C1);
  alpha2_kernel<<<(NN * 32 + 255) / 256, 256, 0, stream>>>(
      h2pre, a_src2, a_dst2, as2, ad2);
  fill_f32<<<(NN + 255) / 256, 256, 0, stream>>>(m2, NEGINF, NN);
  hipMemsetAsync(s2, 0, (size_t)NN * 4, stream);
  hipMemsetAsync(agg2, 0, (size_t)NN * NOUT * 4, stream);
  edge_logits<<<(EPRIME + 255) / 256, 256, 0, stream>>>(ei, as2, ad2, e2b, m2, 1);
  edge_exp<<<(EPRIME + 255) / 256, 256, 0, stream>>>(ei, e2b, m2, s2, 1);
  edge_agg2<<<(EPRIME + 7) / 8, 256, 0, stream>>>(ei, e2b, s2, h2pre, agg2);
  finalize2<<<(NN * 32 + 255) / 256, 256, 0, stream>>>(agg2, b2, out_emb, datan);

  // ---- soft k-means: 10 + 1 update steps ----
  hipMemcpyAsync(mu, mu0, (size_t)KCLUS * NOUT * 4, hipMemcpyDeviceToDevice, stream);
  for (int it = 0; it < 11; it++) {
    hipMemsetAsync(crw, 0, 512, stream);
    hipMemsetAsync(csum, 0, (size_t)KCLUS * NOUT * 4, stream);
    dist_kernel<<<NN / 32, 128, 0, stream>>>(datan, mu, distw);
    softmax_rows<<<(NN * 32 + 255) / 256, 256, 0, stream>>>(distw, rw, crw);
    rtd_kernel<<<160, 256, 0, stream>>>(rw, datan, csum);
    mu_update<<<(KCLUS * NOUT + 255) / 256, 256, 0, stream>>>(csum, crw, mu);
  }
  // final outputs: mu, dist = data_n @ mu^T, r = softmax(temp*dist)
  hipMemcpyAsync(out_mu, mu, (size_t)KCLUS * NOUT * 4, hipMemcpyDeviceToDevice, stream);
  dist_kernel<<<NN / 32, 128, 0, stream>>>(datan, mu, out_dist);
  softmax_rows<<<(NN * 32 + 255) / 256, 256, 0, stream>>>(out_dist, out_r, nullptr);
}
